// VLTQP_82781199663489
// MI455X (gfx1250) — compile-verified, matrix operand order checked by reading, not run on hardware
//
#include <hip/hip_runtime.h>
#include <hip/hip_bf16.h>

// ---------------------------------------------------------------------------
// CDNA5 (gfx1250): wave32, WMMA bf16 16x16x32 GEMMs, bf16 pre-converted
// operands, double-buffered async global->LDS staging (ASYNCcnt pipelining),
// b128 LDS fragment reads.
// ---------------------------------------------------------------------------

typedef __attribute__((ext_vector_type(16))) __bf16 v16bf;
typedef __attribute__((ext_vector_type(8)))  float  v8f;

union FragBF {
    unsigned short u[16];
    v16bf v;
};

struct alignas(16) H8 { unsigned short s[8]; };   // one 16-byte fragment run
struct alignas(16) U4 { unsigned x, y, z, w; };
struct alignas(8)  U2 { unsigned x, y; };

__device__ __forceinline__ unsigned short f2bf(float f) {
    union { float f; unsigned u; } c; c.f = f;
    unsigned u = c.u;
    u += 0x7FFFu + ((u >> 16) & 1u);      // round-to-nearest-even
    return (unsigned short)(u >> 16);
}
__device__ __forceinline__ float bf2f(unsigned short s) {
    union { unsigned u; float f; } c; c.u = ((unsigned)s) << 16;
    return c.f;
}
// order-preserving float -> uint key (ascending)
__device__ __forceinline__ unsigned fkey(float f) {
    union { float f; unsigned u; } c; c.f = f;
    unsigned u = c.u;
    return (u & 0x80000000u) ? ~u : (u | 0x80000000u);
}

// ISA 16-bit A/B fragment layout: two contiguous runs per lane at
// [half*8 .. +7] and [16+half*8 .. +7] of a 32-wide K chunk.
__device__ __forceinline__ void frag_from_row(const unsigned short* rowbase,
                                              int half, FragBF& f) {
    *(H8*)&f.u[0] = *(const H8*)(rowbase + half * 8);
    *(H8*)&f.u[8] = *(const H8*)(rowbase + 16 + half * 8);
}
// scatter 8 packed bf16 (one U4) with stride (for transposed LDS tiles)
__device__ __forceinline__ void scatter_u4_t(unsigned short* dst, int stride, U4 q) {
    dst[0 * stride] = (unsigned short)(q.x & 0xffff);
    dst[1 * stride] = (unsigned short)(q.x >> 16);
    dst[2 * stride] = (unsigned short)(q.y & 0xffff);
    dst[3 * stride] = (unsigned short)(q.y >> 16);
    dst[4 * stride] = (unsigned short)(q.z & 0xffff);
    dst[5 * stride] = (unsigned short)(q.z >> 16);
    dst[6 * stride] = (unsigned short)(q.w & 0xffff);
    dst[7 * stride] = (unsigned short)(q.w >> 16);
}

__device__ __forceinline__ float wred_sum(float v) {
    #pragma unroll
    for (int o = 16; o > 0; o >>= 1) v += __shfl_xor(v, o, 32);
    return v;
}
__device__ __forceinline__ float wred_max(float v) {
    #pragma unroll
    for (int o = 16; o > 0; o >>= 1) v = fmaxf(v, __shfl_xor(v, o, 32));
    return v;
}
__device__ __forceinline__ int wred_isum(int v) {
    #pragma unroll
    for (int o = 16; o > 0; o >>= 1) v += __shfl_xor(v, o, 32);
    return v;
}

// ---------------------------------------------------------------------------
// fp32 -> bf16 bulk conversion (one float4 -> 8B packed per thread)
// ---------------------------------------------------------------------------
__global__ __launch_bounds__(256) void cvt_f32_bf16(
        const float* __restrict__ src, unsigned short* __restrict__ dst, int n4) {
    int i = blockIdx.x * 256 + threadIdx.x;
    if (i >= n4) return;
    float4 v = *(const float4*)(src + (size_t)i * 4);
    U2 o;
    o.x = (unsigned)f2bf(v.x) | ((unsigned)f2bf(v.y) << 16);
    o.y = (unsigned)f2bf(v.z) | ((unsigned)f2bf(v.w) << 16);
    *(U2*)(dst + (size_t)i * 4) = o;
}

// ---------------------------------------------------------------------------
// Double-buffered WMMA GEMM, bf16 operands: C[M,N] = A[M,K] @ B[K,N] + bias.
// Block: 256 threads = 8 waves (2x4); block tile 32x128; K chunk 32.
// Chunk i+1 is staged (A via async global->LDS DMA, B via transposed stores)
// into the alternate buffer while chunk i computes; ASYNCcnt wait lands
// after the WMMAs so the DMA overlaps compute. One barrier per chunk.
// Each wave computes two 16x16 n-tiles sharing one A fragment.
// ---------------------------------------------------------------------------
template <bool OUT_BF16>
__global__ __launch_bounds__(256) void wmma_gemm_bf16(
        const unsigned short* __restrict__ A, const unsigned short* __restrict__ B,
        const float* __restrict__ bias, void* __restrict__ Cout,
        int M, int N, int K) {
    __shared__ unsigned short sA[2][32 * 32];
    __shared__ unsigned short sBt[2][128 * 32];

    int tid = threadIdx.x, wid = tid >> 5, lane = tid & 31;
    int half = lane >> 4, mn = lane & 15;
    int wm = wid >> 2, wn = wid & 3;            // wave grid 2 x 4
    int bm = blockIdx.y, bn = blockIdx.x;

    // A staging (async copy, threads 0..127): 16B each, rows clamped to M-1.
    // Clamped garbage only feeds output rows dropped by the store guard.
    int ar = tid >> 2, ac8 = (tid & 3) * 8;
    int garow = bm * 32 + ar;
    int garc = (garow < M) ? garow : (M - 1);
    const unsigned short* gaBase = A + (size_t)garc * K + ac8;
    unsigned lds_a0 = (unsigned)(unsigned long long)(void*)&sA[0][ar * 32 + ac8];
    unsigned lds_a1 = (unsigned)(unsigned long long)(void*)&sA[1][ar * 32 + ac8];
    // B staging (transposed): row brr, cols bc16..bc16+15 (two U4 per thread)
    int brr = tid >> 3, bc16 = (tid & 7) * 16;
    const unsigned short* gbBase = B + (size_t)brr * N + bn * 128 + bc16;

    const int nch = K >> 5;

    // ---- prologue: stage chunk 0 into buffer 0 ----
    if (tid < 128)
        asm volatile("global_load_async_to_lds_b128 %0, %1, off"
                     :: "v"(lds_a0), "v"(gaBase) : "memory");
    {
        U4 b0 = *(const U4*)(gbBase);
        U4 b1 = *(const U4*)(gbBase + 8);
        scatter_u4_t(&sBt[0][bc16 * 32 + brr], 32, b0);
        scatter_u4_t(&sBt[0][(bc16 + 8) * 32 + brr], 32, b1);
    }
    asm volatile("s_wait_asynccnt 0" ::: "memory");
    __syncthreads();

    v8f acc0 = {}, acc1 = {};
    for (int i = 0; i < nch; ++i) {
        int cur = i & 1, nxt = cur ^ 1;
        if (i + 1 < nch) {
            if (tid < 128) {
                const unsigned short* ga = gaBase + (i + 1) * 32;
                unsigned lds = nxt ? lds_a1 : lds_a0;
                asm volatile("global_load_async_to_lds_b128 %0, %1, off"
                             :: "v"(lds), "v"(ga) : "memory");
            }
            const unsigned short* gb = gbBase + (size_t)(i + 1) * 32 * N;
            U4 b0 = *(const U4*)(gb);
            U4 b1 = *(const U4*)(gb + 8);
            scatter_u4_t(&sBt[nxt][bc16 * 32 + brr], 32, b0);
            scatter_u4_t(&sBt[nxt][(bc16 + 8) * 32 + brr], 32, b1);
        }
        FragBF a, bf0, bf1;
        frag_from_row(&sA[cur][(wm * 16 + mn) * 32], half, a);
        frag_from_row(&sBt[cur][(wn * 16 + mn) * 32], half, bf0);
        frag_from_row(&sBt[cur][(64 + wn * 16 + mn) * 32], half, bf1);
        acc0 = __builtin_amdgcn_wmma_f32_16x16x32_bf16(
                   false, a.v, false, bf0.v, (short)0, acc0, false, false);
        acc1 = __builtin_amdgcn_wmma_f32_16x16x32_bf16(
                   false, a.v, false, bf1.v, (short)0, acc1, false, false);
        asm volatile("s_wait_asynccnt 0" ::: "memory");   // DMA overlapped compute
        __syncthreads();
    }
    #pragma unroll
    for (int r = 0; r < 8; ++r) {
        int m = bm * 32 + wm * 16 + r + half * 8;
        int n0 = bn * 128 + wn * 16 + mn;
        int n1 = n0 + 64;
        if (m < M) {
            float v0 = acc0[r], v1 = acc1[r];
            if (bias) { v0 += bias[n0]; v1 += bias[n1]; }
            if (OUT_BF16) {
                ((unsigned short*)Cout)[(size_t)m * N + n0] = f2bf(v0);
                ((unsigned short*)Cout)[(size_t)m * N + n1] = f2bf(v1);
            } else {
                ((float*)Cout)[(size_t)m * N + n0] = v0;
                ((float*)Cout)[(size_t)m * N + n1] = v1;
            }
        }
    }
}

// ---------------------------------------------------------------------------
// Fused attention (one wave / 16-query tile of one (b,h)), bf16 qkv:
//   S = Q K^T (WMMA) -> LDS strip, per-row top-256 via key bisection,
//   softmax -> implicit scatter+mask -> second softmax, O = W V (WMMA).
//   Output X written as bf16 (only ever consumed as a WMMA A operand).
// ---------------------------------------------------------------------------
#define QKV_AT(b, n, c3, h, d) ((((size_t)(b) * 1024 + (n)) * 3 + (c3)) * 16 + (h)) * 64 + (d)

__global__ __launch_bounds__(32) void attn_fused(
        const unsigned short* __restrict__ qkv,
        const unsigned char* __restrict__ mask,
        unsigned short* __restrict__ X) {
    const int N = 1024, H = 16;
    const float scale = 0.125f;                 // 64^-0.5
    int qb = blockIdx.x & 63;
    int bh = blockIdx.x >> 6;
    int h  = bh % H, b = bh / H;
    int q0 = qb * 16;

    int lane = threadIdx.x & 31;
    int half = lane >> 4, mn = lane & 15;

    __shared__ unsigned short sQ[16 * 64];
    __shared__ unsigned short sS[16 * 1024];    // scores, then weights (bf16)
    __shared__ unsigned short sVt[64 * 32];     // V chunk, transposed

    // stage Q tile (16x64 bf16): pure 16B copies
    #pragma unroll
    for (int j = 0; j < 4; ++j) {
        int f = j * 32 + lane;
        int r = f >> 3, c8 = (f & 7) * 8;
        *(H8*)&sQ[r * 64 + c8] = *(const H8*)(qkv + QKV_AT(b, q0 + r, 0, h, c8));
    }
    __syncthreads();

    // ---- scores: 64 key tiles, K-dim 64 = 2 WMMA chunks ----
    for (int kb = 0; kb < 64; ++kb) {
        const unsigned short* kp = qkv + QKV_AT(b, kb * 16 + mn, 1, h, 0);
        v8f acc = {};
        #pragma unroll
        for (int c = 0; c < 2; ++c) {
            FragBF a, bf;
            frag_from_row(sQ + mn * 64 + 32 * c, half, a);
            frag_from_row(kp + 32 * c, half, bf);
            acc = __builtin_amdgcn_wmma_f32_16x16x32_bf16(
                      false, a.v, false, bf.v, (short)0, acc, false, false);
        }
        #pragma unroll
        for (int r = 0; r < 8; ++r)
            sS[(r + 8 * half) * 1024 + kb * 16 + mn] = f2bf(acc[r] * scale);
    }
    __syncthreads();

    // ---- per-row: top-256 + softmax + mask + second softmax ----
    for (int q = 0; q < 16; ++q) {
        float sv[32];
        unsigned mbits = 0;
        #pragma unroll
        for (int i = 0; i < 32; ++i) {
            int j = i * 32 + lane;
            sv[i] = bf2f(sS[q * 1024 + j]);
            if (mask[(size_t)b * N + j]) mbits |= (1u << i);
        }
        unsigned thr = 0;                       // largest thr: count(>=thr) >= 256
        for (int bit = 31; bit >= 0; --bit) {
            unsigned cand = thr | (1u << bit);
            int c = 0;
            #pragma unroll
            for (int i = 0; i < 32; ++i) c += (fkey(sv[i]) >= cand) ? 1 : 0;
            if (wred_isum(c) >= 256) thr = cand;
        }
        float m = -1e30f;
        #pragma unroll
        for (int i = 0; i < 32; ++i)
            if (fkey(sv[i]) >= thr) m = fmaxf(m, sv[i]);
        m = wred_max(m);
        float s1 = 0.f;
        #pragma unroll
        for (int i = 0; i < 32; ++i)
            if (fkey(sv[i]) >= thr) s1 += __expf(sv[i] - m);
        s1 = wred_sum(s1);
        float inv_s1 = 1.f / s1;
        #pragma unroll
        for (int i = 0; i < 32; ++i)
            sv[i] = (fkey(sv[i]) >= thr) ? __expf(sv[i] - m) * inv_s1 : 0.f;
        float m2 = -1e30f;
        #pragma unroll
        for (int i = 0; i < 32; ++i)
            if (!(mbits & (1u << i))) m2 = fmaxf(m2, sv[i]);
        m2 = wred_max(m2);
        float d2 = 0.f;
        #pragma unroll
        for (int i = 0; i < 32; ++i)
            if (!(mbits & (1u << i))) d2 += __expf(sv[i] - m2);
        d2 = wred_sum(d2);
        float inv_d2 = 1.f / d2;
        #pragma unroll
        for (int i = 0; i < 32; ++i) {
            int j = i * 32 + lane;
            float w = (mbits & (1u << i)) ? 0.f : __expf(sv[i] - m2) * inv_d2;
            sS[q * 1024 + j] = f2bf(w);
        }
    }
    __syncthreads();

    // ---- O = W(16x1024) @ V(1024x64): stage V chunk transposed, 4 n-tiles ----
    v8f o0 = {}, o1 = {}, o2 = {}, o3 = {};
    for (int kc = 0; kc < 32; ++kc) {
        const unsigned short* vp = qkv + QKV_AT(b, kc * 32 + lane, 2, h, 0);
        #pragma unroll
        for (int c8 = 0; c8 < 64; c8 += 8)
            scatter_u4_t(&sVt[c8 * 32 + lane], 32, *(const U4*)(vp + c8));
        __syncthreads();
        FragBF a;
        frag_from_row(sS + mn * 1024 + kc * 32, half, a);
        FragBF b0, b1, b2, b3;
        frag_from_row(sVt + (0 * 16 + mn) * 32, half, b0);
        frag_from_row(sVt + (1 * 16 + mn) * 32, half, b1);
        frag_from_row(sVt + (2 * 16 + mn) * 32, half, b2);
        frag_from_row(sVt + (3 * 16 + mn) * 32, half, b3);
        o0 = __builtin_amdgcn_wmma_f32_16x16x32_bf16(false, a.v, false, b0.v, (short)0, o0, false, false);
        o1 = __builtin_amdgcn_wmma_f32_16x16x32_bf16(false, a.v, false, b1.v, (short)0, o1, false, false);
        o2 = __builtin_amdgcn_wmma_f32_16x16x32_bf16(false, a.v, false, b2.v, (short)0, o2, false, false);
        o3 = __builtin_amdgcn_wmma_f32_16x16x32_bf16(false, a.v, false, b3.v, (short)0, o3, false, false);
        __syncthreads();
    }
    #pragma unroll
    for (int r = 0; r < 8; ++r) {
        int m = r + 8 * half;
        unsigned short* xb = X + ((size_t)b * 1024 + (q0 + m)) * 1024 + h * 64 + mn;
        xb[0]  = f2bf(o0[r]);
        xb[16] = f2bf(o1[r]);
        xb[32] = f2bf(o2[r]);
        xb[48] = f2bf(o3[r]);
    }
}

// ---------------------------------------------------------------------------
// Row LayerNorm: out = LN(x + res) * g + b   (C = 1024, 256 threads/row)
// ---------------------------------------------------------------------------
__global__ void ln_rows(const float* __restrict__ x, const float* __restrict__ res,
                        const float* __restrict__ g, const float* __restrict__ bta,
                        float* __restrict__ out) {
    const int C = 1024;
    int row = blockIdx.x, tid = threadIdx.x;
    __shared__ float s1[256], s2[256];
    float v[4], sum = 0.f, sq = 0.f;
    #pragma unroll
    for (int i = 0; i < 4; ++i) {
        int c = tid + i * 256;
        float t = x[(size_t)row * C + c];
        if (res) t += res[(size_t)row * C + c];
        v[i] = t; sum += t; sq += t * t;
    }
    s1[tid] = sum; s2[tid] = sq;
    __syncthreads();
    for (int o = 128; o > 0; o >>= 1) {
        if (tid < o) { s1[tid] += s1[tid + o]; s2[tid] += s2[tid + o]; }
        __syncthreads();
    }
    float mu  = s1[0] * (1.f / C);
    float var = s2[0] * (1.f / C) - mu * mu;
    float inv = rsqrtf(var + 1e-5f);
    #pragma unroll
    for (int i = 0; i < 4; ++i) {
        int c = tid + i * 256;
        out[(size_t)row * C + c] = (v[i] - mu) * inv * g[c] + bta[c];
    }
}

// ---------------------------------------------------------------------------
// Per-batch: sim = sigmoid(beta + alpha * <l2(vis), l2(tf_n)>), masked -> 0,
// top-256 via bisection, pooled mean of selected tf rows + txt_token.
// ---------------------------------------------------------------------------
__global__ void sim_pool(const float* __restrict__ tf,
                         const float* __restrict__ vis,
                         const float* __restrict__ txt,
                         const unsigned char* __restrict__ mask,
                         const float* __restrict__ alpha,
                         const float* __restrict__ beta,
                         float* __restrict__ tt_pre) {
    const int C = 1024, N = 1024;
    int b = blockIdx.x, tid = threadIdx.x;
    __shared__ float sim[1024];
    __shared__ float sred[256];
    __shared__ int   scnt;

    float acc = 0.f;
    #pragma unroll
    for (int i = 0; i < 4; ++i) {
        float t = vis[(size_t)b * C + tid + i * 256];
        acc += t * t;
    }
    sred[tid] = acc;
    __syncthreads();
    for (int o = 128; o > 0; o >>= 1) {
        if (tid < o) sred[tid] += sred[tid + o];
        __syncthreads();
    }
    float inv_vn = 1.f / fmaxf(sqrtf(sred[0]), 1e-12f);
    __syncthreads();

    float al = alpha[0], be = beta[0];
    for (int n = tid; n < N; n += 256) {
        const float* r = tf + ((size_t)b * N + n) * C;
        const float* w = vis + (size_t)b * C;
        float dot = 0.f, nn = 0.f;
        for (int c = 0; c < C; c += 4) {
            float4 t = *(const float4*)(r + c);
            float4 u = *(const float4*)(w + c);
            dot += t.x * u.x + t.y * u.y + t.z * u.z + t.w * u.w;
            nn  += t.x * t.x + t.y * t.y + t.z * t.z + t.w * t.w;
        }
        float s;
        if (mask[(size_t)b * N + n]) s = 0.f;      // sigmoid(-inf)
        else {
            float z = be + al * dot * inv_vn / fmaxf(sqrtf(nn), 1e-12f);
            s = 1.f / (1.f + __expf(-z));
        }
        sim[n] = s;
    }
    __syncthreads();

    unsigned thr = 0;
    for (int bit = 31; bit >= 0; --bit) {
        unsigned cand = thr | (1u << bit);
        if (tid == 0) scnt = 0;
        __syncthreads();
        int c = 0;
        for (int n = tid; n < N; n += 256)
            if (fkey(sim[n]) >= cand) ++c;
        atomicAdd(&scnt, c);
        __syncthreads();
        if (scnt >= 256) thr = cand;
        __syncthreads();
    }

    #pragma unroll
    for (int i = 0; i < 4; ++i) {
        int c = tid + i * 256;
        float p = 0.f;
        for (int n = 0; n < N; ++n)
            if (fkey(sim[n]) >= thr) p += tf[((size_t)b * N + n) * C + c];
        tt_pre[(size_t)b * C + c] = p * (1.f / 256.f) + txt[(size_t)b * C + c];
    }
}

// ---------------------------------------------------------------------------
extern "C" void kernel_launch(void* const* d_in, const int* in_sizes, int n_in,
                              void* d_out, int out_size, void* d_ws, size_t ws_size,
                              hipStream_t stream) {
    const float* txt   = (const float*)d_in[0];
    const float* tfeat = (const float*)d_in[1];
    const unsigned char* mask = (const unsigned char*)d_in[2];
    const float* vis   = (const float*)d_in[3];
    const float* Wqkv  = (const float*)d_in[5];
    const float* bqkv  = (const float*)d_in[6];
    const float* Wp2   = (const float*)d_in[7];
    const float* bp2   = (const float*)d_in[8];
    const float* ln2g  = (const float*)d_in[9];
    const float* ln2b  = (const float*)d_in[10];
    const float* alpha = (const float*)d_in[11];
    const float* beta  = (const float*)d_in[12];
    const float* Wp1   = (const float*)d_in[13];
    const float* bp1   = (const float*)d_in[14];
    const float* ln1g  = (const float*)d_in[15];
    const float* ln1b  = (const float*)d_in[16];

    float* out = (float*)d_out;
    float* tt = out;               // [8, 1, 1024]
    float* tf = out + 8192;        // [8, 1024, 1024]

    // workspace carve-up
    char* w = (char*)d_ws;
    auto alloc = [&](size_t bytes) -> char* {
        char* p = w; w += (bytes + 255) & ~(size_t)255; return p;
    };
    unsigned short* tfeatH = (unsigned short*)alloc((size_t)8192 * 1024 * 2);
    unsigned short* WqkvH  = (unsigned short*)alloc((size_t)1024 * 3072 * 2);
    unsigned short* Wp2H   = (unsigned short*)alloc((size_t)1024 * 1024 * 2);
    unsigned short* Wp1H   = (unsigned short*)alloc((size_t)1024 * 1024 * 2);
    unsigned short* qkvH   = (unsigned short*)alloc((size_t)8192 * 3072 * 2);
    unsigned short* XH     = (unsigned short*)alloc((size_t)8192 * 1024 * 2);
    unsigned short* ttpreH = (unsigned short*)alloc((size_t)8 * 1024 * 2);
    float* Y     = (float*)alloc((size_t)8192 * 1024 * 4);
    float* ttpre = (float*)alloc((size_t)8 * 1024 * 4);
    float* ttmm  = (float*)alloc((size_t)8 * 1024 * 4);

    // 0. one-time bf16 conversions of all WMMA operands
    cvt_f32_bf16<<<(8192 * 1024 / 4) / 256, 256, 0, stream>>>(tfeat, tfeatH, 8192 * 1024 / 4);
    cvt_f32_bf16<<<(1024 * 3072 / 4) / 256, 256, 0, stream>>>(Wqkv, WqkvH, 1024 * 3072 / 4);
    cvt_f32_bf16<<<(1024 * 1024 / 4) / 256, 256, 0, stream>>>(Wp2, Wp2H, 1024 * 1024 / 4);
    cvt_f32_bf16<<<(1024 * 1024 / 4) / 256, 256, 0, stream>>>(Wp1, Wp1H, 1024 * 1024 / 4);

    // 1. qkv = text_features @ Wqkv + bqkv   (bf16 out)
    wmma_gemm_bf16<true><<<dim3(3072 / 128, 8192 / 32), 256, 0, stream>>>(
        tfeatH, WqkvH, bqkv, qkvH, 8192, 3072, 1024);
    // 2. fused sparse attention -> X (bf16)
    attn_fused<<<8 * 16 * 64, 32, 0, stream>>>(qkvH, mask, XH);
    // 3. Y = X @ Wp2 + bp2  (fp32 out)
    wmma_gemm_bf16<false><<<dim3(1024 / 128, 8192 / 32), 256, 0, stream>>>(
        XH, Wp2H, bp2, Y, 8192, 1024, 1024);
    // 4. tf = LN(text_features + Y)
    ln_rows<<<8192, 256, 0, stream>>>(Y, tfeat, ln2g, ln2b, tf);
    // 5. sim / sigmoid / top-256 / pooled + txt_token
    sim_pool<<<8, 256, 0, stream>>>(tf, vis, txt, mask, alpha, beta, ttpre);
    // 6. ttmm = ttpre @ Wp1 + bp1   (M = 8, row-guarded)
    cvt_f32_bf16<<<(8 * 1024 / 4 + 255) / 256, 256, 0, stream>>>(ttpre, ttpreH, 8 * 1024 / 4);
    wmma_gemm_bf16<false><<<dim3(1024 / 128, 1), 256, 0, stream>>>(
        ttpreH, Wp1H, bp1, ttmm, 8, 1024, 1024);
    // 7. tt = LN(ttmm)
    ln_rows<<<8, 256, 0, stream>>>(ttmm, nullptr, ln1g, ln1b, tt);
}
